// GRUModel_36490042147514
// MI455X (gfx1250) — matrix operand order repaired and checked by reading, not compiled, not run on hardware
//
#include <hip/hip_runtime.h>

// ---------------------------------------------------------------------------
// GRU (4 layers, HID=128, T=512, N=1024) + FC head for MI455X (gfx1250).
// Recurrent and input GEMMs use v_wmma_f32_16x16x32_bf16 (wave32 WMMA).
// Gate nonlinearities use HW fast paths (v_exp_f32 / v_rcp_f32) since the
// 512-step scan is latency-bound and the gate math sits on the critical path.
// ---------------------------------------------------------------------------

typedef __attribute__((ext_vector_type(16))) __bf16 v16bf;
typedef __attribute__((ext_vector_type(8)))  float  v8f;

#define HID 128
#define NB  1024
#define TT  512
#define G3  384   // 3*HID

#define WMMA_BF16(A, B, C) \
  __builtin_amdgcn_wmma_f32_16x16x32_bf16(false, (A), false, (B), (short)0, (C), false, false)

__device__ __forceinline__ unsigned short f2bf(float f) {
  union { float f; unsigned u; } a; a.f = f;
  unsigned r = a.u + 0x7FFFu + ((a.u >> 16) & 1u);   // round-to-nearest-even
  return (unsigned short)(r >> 16);
}
__device__ __forceinline__ float bf2f(unsigned short b) {
  union { unsigned u; float f; } a; a.u = ((unsigned)b) << 16;
  return a.f;
}

// Fast transcendentals: v_exp_f32 computes 2^x; v_rcp_f32 is ~1ulp.
__device__ __forceinline__ float fast_sigmoid(float x) {
  // 1/(1+e^-x) = rcp(1 + 2^(-x*log2(e))).  Saturates correctly at extremes.
  return __builtin_amdgcn_rcpf(1.f + __builtin_amdgcn_exp2f(-1.442695040888963f * x));
}
__device__ __forceinline__ float fast_tanh(float x) {
  // tanh(x) = 2*sigmoid(2x) - 1
  return 2.f * __builtin_amdgcn_rcpf(1.f + __builtin_amdgcn_exp2f(-2.885390081777927f * x)) - 1.f;
}

union Frag {
  unsigned short s[16];
  uint4 q[2];
  v16bf v;
};

// A-matrix 16x32 bf16 fragment gather from a row-major 16xHID bf16 LDS tile.
// Layout: lanes 0-15 hold M=lane, K = {0..15}; lanes 16-31 M=lane-16,
// K = {16..31} (matches load_B). Two aligned 16B LDS loads per lane.
__device__ __forceinline__ v16bf load_A(const unsigned short* tile, int kk, int lane) {
  const int m  = lane & 15;
  const int k0 = kk * 32 + ((lane >> 4) << 4);
  Frag f;
  f.q[0] = *(const uint4*)&tile[m * HID + k0];
  f.q[1] = *(const uint4*)&tile[m * HID + k0 + 8];
  return f.v;
}

// B-matrix 32x16 bf16 fragment: lane = output column (both halves), element e
// maps to K = kbase + e (kbase = kk*32 + 16*(lane>=16)). B[k][n] = W[n][k],
// so per lane this is 16 contiguous f32 of one W row, converted to bf16.
__device__ __forceinline__ v16bf load_B(const float* __restrict__ W, int ldk,
                                        int row, int kbase) {
  Frag f;
  const float* p = W + (size_t)row * ldk + kbase;
#pragma unroll
  for (int e = 0; e < 16; ++e) f.s[e] = f2bf(p[e]);
  return f.v;
}

template <bool FIRST>
__global__ void __launch_bounds__(256)
gru_layer_kernel(const float* __restrict__ x0,            // FIRST: (T, 2, N) f32
                 const unsigned short* __restrict__ xin,  // !FIRST: (T, N, H) bf16
                 const float* __restrict__ W_ih,          // (3H, F)
                 const float* __restrict__ W_hh,          // (3H, H)
                 const float* __restrict__ b_ih,          // (3H,)
                 const float* __restrict__ b_hh,          // (3H,)
                 unsigned short* __restrict__ out)        // (T, N, H) bf16
{
  __shared__ __align__(16) unsigned short hbuf[16 * HID];
  __shared__ __align__(16) unsigned short xbuf[16 * HID];

  const int tid  = threadIdx.x;
  const int lane = tid & 31;
  const int w    = tid >> 5;               // wave 0..7 -> hidden column block
  const int half = lane >> 4;
  const int nb   = blockIdx.x * 16;        // batch-row base of this block
  const int col  = (w << 4) + (lane & 15); // hidden column 0..127 (this lane)

  // ---- Stage W_hh / W_ih B-fragments into registers (invariant over T) ----
  v16bf Bhh[3][4];
  v16bf Bih[3][4];
#pragma unroll
  for (int g = 0; g < 3; ++g) {
    const int row = g * HID + col;
#pragma unroll
    for (int kk = 0; kk < 4; ++kk) {
      const int kbase = kk * 32 + (half << 4);
      Bhh[g][kk] = load_B(W_hh, HID, row, kbase);
      if (!FIRST) Bih[g][kk] = load_B(W_ih, HID, row, kbase);
    }
  }

  const float bih_r = b_ih[0 * HID + col];
  const float bih_z = b_ih[1 * HID + col];
  const float bih_n = b_ih[2 * HID + col];
  const float bhh_r = b_hh[0 * HID + col];
  const float bhh_z = b_hh[1 * HID + col];
  const float bhh_n = b_hh[2 * HID + col];

  // Layer-0 input weights: W_ih0 is (3H, 2)
  float wr0 = 0.f, wr1 = 0.f, wz0 = 0.f, wz1 = 0.f, wn0 = 0.f, wn1 = 0.f;
  if (FIRST) {
    wr0 = W_ih[(0 * HID + col) * 2 + 0]; wr1 = W_ih[(0 * HID + col) * 2 + 1];
    wz0 = W_ih[(1 * HID + col) * 2 + 0]; wz1 = W_ih[(1 * HID + col) * 2 + 1];
    wn0 = W_ih[(2 * HID + col) * 2 + 0]; wn1 = W_ih[(2 * HID + col) * 2 + 1];
  }

  // ---- h = 0 ----
  float hreg[8];
#pragma unroll
  for (int i = 0; i < 8; ++i) hreg[i] = 0.f;
  for (int i = tid; i < 16 * HID; i += 256) hbuf[i] = 0;
  __syncthreads();

  for (int t = 0; t < TT; ++t) {
    // ---- stage x tile (16 x 128 bf16) into LDS ----
    if (!FIRST) {
      const int r  = tid >> 4;
      const int c8 = (tid & 15) << 3;
      *(uint4*)&xbuf[r * HID + c8] =
          *(const uint4*)&xin[(size_t)t * NB * HID + (size_t)(nb + r) * HID + c8];
      __syncthreads();   // xbuf staged AND previous step's hbuf writes visible
    }

    // ---- accumulators (C/D layout: element i -> m = i + 8*half, n = col) ----
    v8f hr, hz, hn, gr, gz, gn;
#pragma unroll
    for (int i = 0; i < 8; ++i) { hr[i] = bhh_r; hz[i] = bhh_z; hn[i] = bhh_n; }

    if (FIRST) {
#pragma unroll
      for (int i = 0; i < 8; ++i) {
        const int m = i + (half << 3);
        const float xa = x0[(size_t)t * 2 * NB + nb + m];
        const float xb = x0[(size_t)t * 2 * NB + NB + nb + m];
        gr[i] = bih_r + xa * wr0 + xb * wr1;
        gz[i] = bih_z + xa * wz0 + xb * wz1;
        gn[i] = bih_n + xa * wn0 + xb * wn1;
      }
    } else {
#pragma unroll
      for (int i = 0; i < 8; ++i) { gr[i] = bih_r; gz[i] = bih_z; gn[i] = bih_n; }
    }

    // ---- WMMA: gh = h @ W_hh^T (+ gi = x @ W_ih^T for layers > 0) ----
#pragma unroll
    for (int kk = 0; kk < 4; ++kk) {
      const v16bf Ah = load_A(hbuf, kk, lane);
      hr = WMMA_BF16(Ah, Bhh[0][kk], hr);
      hz = WMMA_BF16(Ah, Bhh[1][kk], hz);
      hn = WMMA_BF16(Ah, Bhh[2][kk], hn);
      if (!FIRST) {
        const v16bf Ax = load_A(xbuf, kk, lane);
        gr = WMMA_BF16(Ax, Bih[0][kk], gr);
        gz = WMMA_BF16(Ax, Bih[1][kk], gz);
        gn = WMMA_BF16(Ax, Bih[2][kk], gn);
      }
    }

    __syncthreads();  // everyone done reading hbuf/xbuf for this step

    // ---- gates + state update (PyTorch order r, z, n) ----
#pragma unroll
    for (int i = 0; i < 8; ++i) {
      const float r = fast_sigmoid(gr[i] + hr[i]);
      const float z = fast_sigmoid(gz[i] + hz[i]);
      const float n = fast_tanh(gn[i] + r * hn[i]);
      hreg[i] = (1.f - z) * n + z * hreg[i];
    }

    // ---- write back h (bf16) to LDS and layer output ----
#pragma unroll
    for (int i = 0; i < 8; ++i) {
      const int m = i + (half << 3);
      const unsigned short hb = f2bf(hreg[i]);
      hbuf[m * HID + col] = hb;
      out[(size_t)t * NB * HID + (size_t)(nb + m) * HID + col] = hb;
    }
    if (FIRST) __syncthreads();  // hbuf writes before next step's gathers
  }
}

// last = h3[:, N-1, :]  (512 x 128); hidden = relu(last @ fc1^T + b1);
// out = hidden @ fc2^T + b2  -> (512, 3) f32
__global__ void __launch_bounds__(128)
fc_head_kernel(const unsigned short* __restrict__ h3,
               const float* __restrict__ fc1_w, const float* __restrict__ fc1_b,
               const float* __restrict__ fc2_w, const float* __restrict__ fc2_b,
               float* __restrict__ outp)
{
  __shared__ float last[HID];
  __shared__ float hid[HID];
  const int t = blockIdx.x;
  const int j = threadIdx.x;

  last[j] = bf2f(h3[(size_t)t * NB * HID + (size_t)(NB - 1) * HID + j]);
  __syncthreads();

  float s = fc1_b[j];
#pragma unroll 8
  for (int k = 0; k < HID; ++k) s += last[k] * fc1_w[j * HID + k];
  hid[j] = fmaxf(s, 0.f);
  __syncthreads();

  if (j < 3) {
    float s2 = fc2_b[j];
#pragma unroll 8
    for (int k = 0; k < HID; ++k) s2 += hid[k] * fc2_w[j * HID + k];
    outp[t * 3 + j] = s2;
  }
}

extern "C" void kernel_launch(void* const* d_in, const int* in_sizes, int n_in,
                              void* d_out, int out_size, void* d_ws, size_t ws_size,
                              hipStream_t stream) {
  (void)in_sizes; (void)n_in; (void)out_size; (void)ws_size;
  const float* x     = (const float*)d_in[0];   // (512, 2, 1024)
  const float* W_ih0 = (const float*)d_in[1];   // (384, 2)
  const float* W_ihR = (const float*)d_in[2];   // (3, 384, 128)
  const float* W_hh  = (const float*)d_in[3];   // (4, 384, 128)
  const float* b_ih  = (const float*)d_in[4];   // (4, 384)
  const float* b_hh  = (const float*)d_in[5];   // (4, 384)
  const float* fc1_w = (const float*)d_in[6];
  const float* fc1_b = (const float*)d_in[7];
  const float* fc2_w = (const float*)d_in[8];
  const float* fc2_b = (const float*)d_in[9];
  float* outp = (float*)d_out;

  // Ping-pong bf16 activation buffers in workspace: 2 x (512*1024*128) bf16.
  unsigned short* bufA = (unsigned short*)d_ws;
  unsigned short* bufB = bufA + (size_t)TT * NB * HID;

  const dim3 grid(NB / 16), blk(256);

  gru_layer_kernel<true><<<grid, blk, 0, stream>>>(
      x, nullptr, W_ih0, W_hh, b_ih, b_hh, bufA);
  gru_layer_kernel<false><<<grid, blk, 0, stream>>>(
      nullptr, bufA, W_ihR + 0 * G3 * HID, W_hh + 1 * G3 * HID,
      b_ih + 1 * G3, b_hh + 1 * G3, bufB);
  gru_layer_kernel<false><<<grid, blk, 0, stream>>>(
      nullptr, bufB, W_ihR + 1 * G3 * HID, W_hh + 2 * G3 * HID,
      b_ih + 2 * G3, b_hh + 2 * G3, bufA);
  gru_layer_kernel<false><<<grid, blk, 0, stream>>>(
      nullptr, bufA, W_ihR + 2 * G3 * HID, W_hh + 3 * G3 * HID,
      b_ih + 3 * G3, b_hh + 3 * G3, bufB);

  fc_head_kernel<<<dim3(TT), dim3(HID), 0, stream>>>(
      bufB, fc1_w, fc1_b, fc2_w, fc2_b, outp);
}
